// TabNSA_36386962932044
// MI455X (gfx1250) — matrix operand order repaired and checked by reading, not compiled, not run on hardware
//
#include <hip/hip_runtime.h>
#include <hip/hip_bf16.h>
#include <math.h>

typedef __attribute__((ext_vector_type(16))) _Float16 v16h;
typedef __attribute__((ext_vector_type(8)))  float    v8f;
typedef __attribute__((ext_vector_type(4)))  int      v4i;

#define B_   32
#define N_   256
#define D_   64
#define H_   8
#define DH_  16
#define WW_  32
#define BLK_ 16
#define KSEL_ 4
#define FF_  256
#define DI_  128     // H_*DH_
#define NBLK_ 16     // N_/BLK_
#define SCALE_ 0.25f // DH^-0.5

#define GLOBAL_AS __attribute__((address_space(1)))
#define LDS_AS    __attribute__((address_space(3)))

#if __has_builtin(__builtin_amdgcn_global_load_async_to_lds_b128)
#define HAVE_ASYNC_LDS 1
#else
#define HAVE_ASYNC_LDS 0
#endif

__device__ __forceinline__ void async_wait_all() {
#if HAVE_ASYNC_LDS
#if __has_builtin(__builtin_amdgcn_s_wait_asynccnt)
  __builtin_amdgcn_s_wait_asynccnt(0);
#else
  asm volatile("s_wait_asynccnt 0x0" ::: "memory");
#endif
#endif
}

__device__ __forceinline__ float gelu_f(float x) {
  float x3 = x * x * x;
  return 0.5f * x * (1.0f + tanhf(0.7978845608028654f * (x + 0.044715f * x3)));
}

// ---------------------------------------------------------------- f32 -> f16
__global__ void k_cvt16(const float* __restrict__ s, _Float16* __restrict__ d, int n) {
  int i = blockIdx.x * blockDim.x + threadIdx.x;
  if (i < n) d[i] = (_Float16)s[i];
}

// ---------------------------------------------------- embed + rmsnorm + ln1^T
__global__ void k_embed(const float* __restrict__ x, const float* __restrict__ W_emb,
                        const float* __restrict__ b_emb, const float* __restrict__ rms_g,
                        const float* __restrict__ ln1_g, const float* __restrict__ ln1_b,
                        float* __restrict__ emb, _Float16* __restrict__ h16,
                        _Float16* __restrict__ ln1t16) {
  __shared__ float red[D_];
  int row = blockIdx.x;   // b*N + n
  int d   = threadIdx.x;  // 0..63
  float e = x[row] * W_emb[d] + b_emb[d];
  emb[(size_t)row * D_ + d] = e;

  red[d] = e; __syncthreads();
  for (int s = D_ / 2; s > 0; s >>= 1) { if (d < s) red[d] += red[d + s]; __syncthreads(); }
  float sum = red[0]; __syncthreads();
  red[d] = e * e; __syncthreads();
  for (int s = D_ / 2; s > 0; s >>= 1) { if (d < s) red[d] += red[d + s]; __syncthreads(); }
  float sumsq = red[0];

  float m   = sum * (1.0f / D_);
  float msq = sumsq * (1.0f / D_);
  float var = msq - m * m;

  float hn = e * rsqrtf(msq + 1e-6f) * rms_g[d];
  h16[(size_t)row * D_ + d] = (_Float16)hn;

  float l1 = (e - m) * rsqrtf(var + 1e-5f) * ln1_g[d] + ln1_b[d];
  int b = row / N_, n = row % N_;
  ln1t16[((size_t)(b * D_ + d)) * N_ + n] = (_Float16)l1;
}

// ---------------------------------------------------------------- gates
__global__ void k_gates(const _Float16* __restrict__ h16, const float* __restrict__ W_g,
                        const float* __restrict__ b_g, float* __restrict__ g) {
  int idx = blockIdx.x * blockDim.x + threadIdx.x;
  if (idx >= B_ * N_ * 3 * H_) return;
  int row = idx / (3 * H_);
  int j   = idx % (3 * H_);
  const _Float16* hr = h16 + (size_t)row * D_;
  float acc = b_g[j];
  for (int dd = 0; dd < D_; ++dd) acc += (float)hr[dd] * W_g[dd * (3 * H_) + j];
  g[idx] = 1.0f / (1.0f + expf(-acc));
}

// ===================== 4-wave tiled WMMA GEMM (64x64 block tile) ============
// block = 128 threads (4 waves). grid = (M/64, Nout/64, batch).
// Wave w computes rows [tm*64+w*16, +16) x cols [tn*64, +64): 4 WMMA accums.
// A chunk (64x32 f16) staged to LDS via async global->LDS (row-major).
// B chunk (32x64 f16) staged to LDS *transposed* (column-major) so each
// fragment is 16 contiguous halves -> ds_load_b128.
// mode: 0 = store f32 raw; 1 = bias+gelu -> f16; 2 = bias -> f32
__global__ void k_wmma_gemm_t64(const _Float16* __restrict__ A,
                                const _Float16* __restrict__ Bm,
                                float* __restrict__ C, _Float16* __restrict__ C16,
                                const float* __restrict__ bias,
                                int M, int K, int Nout,
                                unsigned long long sA, unsigned long long sB,
                                unsigned long long sC, int mode) {
  __shared__ _Float16 ldsA[64 * 32];   // [row][k]   row-major, 4 KB
  __shared__ _Float16 ldsB[64 * 32];   // [n][k]     column-major(T), 4 KB

  int t    = threadIdx.x;        // 0..127
  int w    = t >> 5;             // wave id 0..3
  int lane = t & 31;
  int half = lane >> 4;
  int mloc = lane & 15;
  int tm = blockIdx.x, tn = blockIdx.y, bz = blockIdx.z;

  const _Float16* Abase = A  + (size_t)bz * sA;
  const _Float16* Bbase = Bm + (size_t)bz * sB;

  // staging assignments
  int ar    = t >> 1;            // A: row 0..63
  int apart = t & 1;             // A: 16-half (32B) part within chunk row
  int bk    = t >> 2;            // B: k row 0..31
  int bgrp  = t & 3;             // B: 16-col group

  v8f c0 = {}, c1 = {}, c2 = {}, c3 = {};

  for (int kk = 0; kk < K; kk += 32) {
    // ---- stage A chunk: rows tm*64..+64, k kk..kk+32 (row-major in LDS)
    {
      const _Float16* gp = Abase + (size_t)(tm * 64 + ar) * K + kk + apart * 16;
      _Float16* lp = &ldsA[ar * 32 + apart * 16];
#if HAVE_ASYNC_LDS
      __builtin_amdgcn_global_load_async_to_lds_b128(
          (GLOBAL_AS v4i*)gp, (LDS_AS v4i*)lp, 0, 0);
      __builtin_amdgcn_global_load_async_to_lds_b128(
          (GLOBAL_AS v4i*)(gp + 8), (LDS_AS v4i*)(lp + 8), 0, 0);
#else
      *(float4*)lp = *(const float4*)gp;
      *(float4*)(lp + 8) = *(const float4*)(gp + 8);
#endif
    }
    // ---- stage B chunk transposed: ldsB[n][k] = B[kk+k][tn*64+n]
    {
      const _Float16* gp = Bbase + (size_t)(kk + bk) * Nout + tn * 64 + bgrp * 16;
      _Float16 tmp[16];
#pragma unroll
      for (int i = 0; i < 16; ++i) tmp[i] = gp[i];      // 2x b128 coalesced
#pragma unroll
      for (int i = 0; i < 16; ++i) ldsB[(bgrp * 16 + i) * 32 + bk] = tmp[i];
    }
    if (kk + 32 < K) {
      __builtin_prefetch(Abase + (size_t)(tm * 64 + ar) * K + kk + 32, 0, 1);
      __builtin_prefetch(Bbase + (size_t)(kk + 32 + bk) * Nout + tn * 64, 0, 1);
    }
    async_wait_all();
    __syncthreads();

    // ---- A fragment for this wave (two contiguous 16B LDS reads)
    v16h a;
    {
      const _Float16* arow = &ldsA[(w * 16 + mloc) * 32];
#pragma unroll
      for (int i = 0; i < 8; ++i) a[i]     = arow[half * 16 + i];
#pragma unroll
      for (int i = 0; i < 8; ++i) a[8 + i] = arow[half * 16 + 8 + i];
    }
    // ---- 4 B fragments + 4 WMMAs
    {
      v16h b;
      const _Float16* br;
      br = &ldsB[(0 * 16 + mloc) * 32 + half * 16];
#pragma unroll
      for (int j = 0; j < 16; ++j) b[j] = br[j];
      c0 = __builtin_amdgcn_wmma_f32_16x16x32_f16(false, a, false, b, (short)0, c0, false, false);
      br = &ldsB[(1 * 16 + mloc) * 32 + half * 16];
#pragma unroll
      for (int j = 0; j < 16; ++j) b[j] = br[j];
      c1 = __builtin_amdgcn_wmma_f32_16x16x32_f16(false, a, false, b, (short)0, c1, false, false);
      br = &ldsB[(2 * 16 + mloc) * 32 + half * 16];
#pragma unroll
      for (int j = 0; j < 16; ++j) b[j] = br[j];
      c2 = __builtin_amdgcn_wmma_f32_16x16x32_f16(false, a, false, b, (short)0, c2, false, false);
      br = &ldsB[(3 * 16 + mloc) * 32 + half * 16];
#pragma unroll
      for (int j = 0; j < 16; ++j) b[j] = br[j];
      c3 = __builtin_amdgcn_wmma_f32_16x16x32_f16(false, a, false, b, (short)0, c3, false, false);
    }
    __syncthreads();
  }

  // ---- epilogue
  int rowb = tm * 64 + w * 16 + half * 8;
#pragma unroll
  for (int cidx = 0; cidx < 4; ++cidx) {
    v8f acc = (cidx == 0) ? c0 : (cidx == 1) ? c1 : (cidx == 2) ? c2 : c3;
    int col = tn * 64 + cidx * 16 + mloc;
    float bv = bias ? bias[col] : 0.0f;
#pragma unroll
    for (int r = 0; r < 8; ++r) {
      size_t off = (size_t)bz * sC + (size_t)(rowb + r) * Nout + col;
      float v = acc[r] + bv;
      if (mode == 1)      C16[off] = (_Float16)gelu_f(v);
      else if (mode == 2) C[off]   = v;
      else                C[off]   = acc[r];
    }
  }
}

// ================= 1-wave WMMA GEMM (for small Nout=16 GEMMs) ==============
__global__ void k_wmma_gemm(const _Float16* __restrict__ A, const _Float16* __restrict__ Bm,
                            float* __restrict__ C,
                            int M, int K, int Nout,
                            unsigned long long sA, unsigned long long sB,
                            unsigned long long sC) {
  int lane = threadIdx.x;
  int half = lane >> 4;
  int mloc = lane & 15;
  int tm = blockIdx.x, tn = blockIdx.y, bz = blockIdx.z;

  const _Float16* Ab = A  + (size_t)bz * sA + (size_t)(tm * 16 + mloc) * K;
  const _Float16* Bb = Bm + (size_t)bz * sB + (tn * 16 + mloc);

  v8f c = {};
  for (int kk = 0; kk < K; kk += 32) {
    v16h a, b;
    int abase = kk + half * 16;
#pragma unroll
    for (int i = 0; i < 8; ++i) a[i]     = Ab[abase + i];
#pragma unroll
    for (int i = 0; i < 8; ++i) a[8 + i] = Ab[abase + 8 + i];
    int bbase = kk + half * 16;
#pragma unroll
    for (int j = 0; j < 16; ++j) b[j] = Bb[(size_t)(bbase + j) * Nout];
    c = __builtin_amdgcn_wmma_f32_16x16x32_f16(false, a, false, b, (short)0, c, false, false);
  }
  int col = tn * 16 + mloc;
#pragma unroll
  for (int r = 0; r < 8; ++r) {
    int row = tm * 16 + half * 8 + r;
    C[(size_t)bz * sC + (size_t)row * Nout + col] = c[r];
  }
}

// --------------------------------------------- build per-head (kb, vb) f16 tiles
__global__ void k_build_kvb(const float* __restrict__ qkv, const float* __restrict__ k_pos,
                            const float* __restrict__ v_pos,
                            _Float16* __restrict__ kb16, _Float16* __restrict__ vb16) {
  int idx = blockIdx.x * blockDim.x + threadIdx.x;
  if (idx >= H_ * B_ * NBLK_ * BLK_ * DH_) return;
  int c    = idx % (BLK_ * DH_);
  int brow = (idx / (BLK_ * DH_)) % (B_ * NBLK_);
  int h    = idx / (BLK_ * DH_ * B_ * NBLK_);
  int bl = c / DH_, d = c % DH_;
  int b = brow / NBLK_, nb = brow % NBLK_;
  int n = nb * BLK_ + bl;
  size_t qoff = ((size_t)(b * N_ + n)) * (3 * DI_);
  float kv = qkv[qoff + DI_     + h * DH_ + d] + k_pos[c];
  float vv = qkv[qoff + 2 * DI_ + h * DH_ + d] + v_pos[c];
  size_t o = ((size_t)h * (B_ * NBLK_) + brow) * (BLK_ * DH_) + c;
  kb16[o] = (_Float16)kv;
  vb16[o] = (_Float16)vv;
}

// ------------------------------------ kc/vc: add bias, prepend memory token
__global__ void k_finish_kv(const float* __restrict__ kcg, const float* __restrict__ vcg,
                            const float* __restrict__ b_kc, const float* __restrict__ b_vc,
                            const float* __restrict__ mem_k, const float* __restrict__ mem_v,
                            float* __restrict__ kcf, float* __restrict__ vcf) {
  int idx = blockIdx.x * blockDim.x + threadIdx.x;
  if (idx >= B_ * H_ * (NBLK_ + 1) * DH_) return;
  int d = idx % DH_;
  int j = (idx / DH_) % (NBLK_ + 1);
  int h = (idx / (DH_ * (NBLK_ + 1))) % H_;
  int b = idx / (DH_ * (NBLK_ + 1) * H_);
  float kvl, vvl;
  if (j == 0) { kvl = mem_k[h * DH_ + d]; vvl = mem_v[h * DH_ + d]; }
  else {
    size_t go = ((size_t)h * (B_ * NBLK_) + b * NBLK_ + (j - 1)) * DH_ + d;
    kvl = kcg[go] + b_kc[h * DH_ + d];
    vvl = vcg[go] + b_vc[h * DH_ + d];
  }
  kcf[idx] = kvl; vcf[idx] = vvl;   // layout ((b*H+h)*17 + j)*16 + d
}

// ------------------------------------------------ fused 3-branch attention
__global__ void k_attn(const float* __restrict__ qkv, const float* __restrict__ kcf,
                       const float* __restrict__ vcf, const float* __restrict__ g,
                       _Float16* __restrict__ o16) {
  int idx = blockIdx.x * blockDim.x + threadIdx.x;
  if (idx >= B_ * H_ * N_) return;
  int n = idx % N_;
  int h = (idx / N_) % H_;
  int b = idx / (N_ * H_);

  const float* qp = qkv + ((size_t)(b * N_ + n)) * (3 * DI_) + h * DH_;
  float q[DH_];
#pragma unroll
  for (int d = 0; d < DH_; ++d) q[d] = qp[d];

  // ---- compressed-context attention (17 keys incl. memory token)
  const float* kcb = kcf + ((size_t)(b * H_ + h)) * (NBLK_ + 1) * DH_;
  const float* vcb = vcf + ((size_t)(b * H_ + h)) * (NBLK_ + 1) * DH_;
  float p[NBLK_ + 1];
  float mx = -3.0e38f;
  for (int j = 0; j < NBLK_ + 1; ++j) {
    float s = 0.f;
#pragma unroll
    for (int d = 0; d < DH_; ++d) s += q[d] * kcb[j * DH_ + d];
    s *= SCALE_;
    p[j] = s; if (s > mx) mx = s;
  }
  float den = 0.f;
  for (int j = 0; j < NBLK_ + 1; ++j) { p[j] = expf(p[j] - mx); den += p[j]; }
  float inv = 1.0f / den;
  float out_c[DH_];
#pragma unroll
  for (int d = 0; d < DH_; ++d) out_c[d] = 0.f;
  for (int j = 0; j < NBLK_ + 1; ++j)
#pragma unroll
    for (int d = 0; d < DH_; ++d) out_c[d] += p[j] * vcb[j * DH_ + d];
#pragma unroll
  for (int d = 0; d < DH_; ++d) out_c[d] *= inv;

  // ---- top-KSEL block selection on pc[...,1:]
  int sel[KSEL_];
  bool taken[NBLK_];
#pragma unroll
  for (int j = 0; j < NBLK_; ++j) taken[j] = false;
  for (int t = 0; t < KSEL_; ++t) {
    int best = 0; float bv = -3.0e38f;
    for (int j = 0; j < NBLK_; ++j)
      if (!taken[j] && p[j + 1] > bv) { bv = p[j + 1]; best = j; }
    taken[best] = true; sel[t] = best;
  }

  // ---- fine attention over 4 selected blocks (64 keys), two-pass softmax
  float m2 = -3.0e38f;
  for (int t = 0; t < KSEL_; ++t)
    for (int l = 0; l < BLK_; ++l) {
      int n2 = sel[t] * BLK_ + l;
      const float* kp = qkv + ((size_t)(b * N_ + n2)) * (3 * DI_) + DI_ + h * DH_;
      float s = 0.f;
#pragma unroll
      for (int d = 0; d < DH_; ++d) s += q[d] * kp[d];
      s *= SCALE_;
      if (s > m2) m2 = s;
    }
  float out_f[DH_]; float den2 = 0.f;
#pragma unroll
  for (int d = 0; d < DH_; ++d) out_f[d] = 0.f;
  for (int t = 0; t < KSEL_; ++t)
    for (int l = 0; l < BLK_; ++l) {
      int n2 = sel[t] * BLK_ + l;
      const float* kp = qkv + ((size_t)(b * N_ + n2)) * (3 * DI_) + DI_ + h * DH_;
      const float* vp = kp + DI_;
      float s = 0.f;
#pragma unroll
      for (int d = 0; d < DH_; ++d) s += q[d] * kp[d];
      float e = expf(s * SCALE_ - m2);
      den2 += e;
#pragma unroll
      for (int d = 0; d < DH_; ++d) out_f[d] += e * vp[d];
    }
  float inv2 = 1.0f / den2;
#pragma unroll
  for (int d = 0; d < DH_; ++d) out_f[d] *= inv2;

  // ---- causal sliding-window attention (W=32), two-pass softmax
  float m3 = -3.0e38f;
  for (int w = 0; w < WW_; ++w) {
    int n2 = n - w; if (n2 < 0) break;
    const float* kp = qkv + ((size_t)(b * N_ + n2)) * (3 * DI_) + DI_ + h * DH_;
    float s = 0.f;
#pragma unroll
    for (int d = 0; d < DH_; ++d) s += q[d] * kp[d];
    s *= SCALE_;
    if (s > m3) m3 = s;
  }
  float out_s[DH_]; float den3 = 0.f;
#pragma unroll
  for (int d = 0; d < DH_; ++d) out_s[d] = 0.f;
  for (int w = 0; w < WW_; ++w) {
    int n2 = n - w; if (n2 < 0) break;
    const float* kp = qkv + ((size_t)(b * N_ + n2)) * (3 * DI_) + DI_ + h * DH_;
    const float* vp = kp + DI_;
    float s = 0.f;
#pragma unroll
    for (int d = 0; d < DH_; ++d) s += q[d] * kp[d];
    float e = expf(s * SCALE_ - m3);
    den3 += e;
#pragma unroll
    for (int d = 0; d < DH_; ++d) out_s[d] += e * vp[d];
  }
  float inv3 = 1.0f / den3;

  // ---- gated merge
  const float* gp = g + ((size_t)(b * N_ + n)) * (3 * H_);
  float gc = gp[h], gf = gp[H_ + h], gs = gp[2 * H_ + h];
  _Float16* op = o16 + ((size_t)(b * N_ + n)) * DI_ + h * DH_;
#pragma unroll
  for (int d = 0; d < DH_; ++d) {
    float o = gc * out_c[d] + gf * out_f[d] + gs * (out_s[d] * inv3);
    op[d] = (_Float16)o;
  }
}

// --------------------------- y = emb + t^T ; f = layernorm(y) (f16 for GEMM)
__global__ void k_make_y(const float* __restrict__ emb, const float* __restrict__ t2f,
                         const float* __restrict__ ln2_g, const float* __restrict__ ln2_b,
                         float* __restrict__ y, _Float16* __restrict__ fn16) {
  __shared__ float red[D_];
  int row = blockIdx.x;   // b*N + n
  int d   = threadIdx.x;
  int b = row / N_, n = row % N_;
  float v = emb[(size_t)row * D_ + d] + t2f[((size_t)(b * D_ + d)) * N_ + n];
  y[(size_t)row * D_ + d] = v;

  red[d] = v; __syncthreads();
  for (int s = D_ / 2; s > 0; s >>= 1) { if (d < s) red[d] += red[d + s]; __syncthreads(); }
  float sum = red[0]; __syncthreads();
  red[d] = v * v; __syncthreads();
  for (int s = D_ / 2; s > 0; s >>= 1) { if (d < s) red[d] += red[d + s]; __syncthreads(); }
  float sumsq = red[0];

  float m = sum * (1.0f / D_);
  float var = sumsq * (1.0f / D_) - m * m;
  fn16[(size_t)row * D_ + d] = (_Float16)((v - m) * rsqrtf(var + 1e-5f) * ln2_g[d] + ln2_b[d]);
}

// ------------------------------------- pooled = mean_n(attn_out + y + ff)
__global__ void k_pool(const float* __restrict__ y, const float* __restrict__ ff,
                       const float* __restrict__ attnout, float* __restrict__ pooled) {
  int idx = blockIdx.x * blockDim.x + threadIdx.x;
  if (idx >= B_ * D_) return;
  int b = idx / D_, d = idx % D_;
  float acc = 0.f;
  for (int n = 0; n < N_; ++n) {
    size_t o = ((size_t)(b * N_ + n)) * D_ + d;
    acc += y[o] + ff[o] + attnout[o];
  }
  pooled[idx] = acc * (1.0f / N_);
}

// ------------------------------------------------------------- output head
__global__ void k_head(const float* __restrict__ pooled, const float* __restrict__ W_h1,
                       const float* __restrict__ b_h1, const float* __restrict__ W_h2,
                       const float* __restrict__ b_h2, float* __restrict__ out) {
  __shared__ float hbuf[32];
  int b = blockIdx.x;
  int j = threadIdx.x;   // 32 threads
  float acc = b_h1[j];
  for (int d = 0; d < D_; ++d) acc += pooled[b * D_ + d] * W_h1[d * 32 + j];
  hbuf[j] = gelu_f(acc);
  __syncthreads();
  if (j < 2) {
    float o = b_h2[j];
    for (int t = 0; t < 32; ++t) o += hbuf[t] * W_h2[t * 2 + j];
    out[b * 2 + j] = o;
  }
}

// ===========================================================================
extern "C" void kernel_launch(void* const* d_in, const int* in_sizes, int n_in,
                              void* d_out, int out_size, void* d_ws, size_t ws_size,
                              hipStream_t stream) {
  const float* x     = (const float*)d_in[0];
  const float* W_emb = (const float*)d_in[1];
  const float* b_emb = (const float*)d_in[2];
  const float* rms_g = (const float*)d_in[3];
  const float* W_qkv = (const float*)d_in[4];
  const float* k_pos = (const float*)d_in[5];
  const float* v_pos = (const float*)d_in[6];
  const float* W_kc  = (const float*)d_in[7];
  const float* b_kc  = (const float*)d_in[8];
  const float* W_vc  = (const float*)d_in[9];
  const float* b_vc  = (const float*)d_in[10];
  const float* mem_k = (const float*)d_in[11];
  const float* mem_v = (const float*)d_in[12];
  const float* W_g   = (const float*)d_in[13];
  const float* b_g   = (const float*)d_in[14];
  const float* W_o   = (const float*)d_in[15];
  const float* ln1_g = (const float*)d_in[16];
  const float* ln1_b = (const float*)d_in[17];
  const float* W_t1  = (const float*)d_in[18];
  const float* b_t1  = (const float*)d_in[19];
  const float* W_t2  = (const float*)d_in[20];
  const float* b_t2  = (const float*)d_in[21];
  const float* ln2_g = (const float*)d_in[22];
  const float* ln2_b = (const float*)d_in[23];
  const float* W_f1  = (const float*)d_in[24];
  const float* b_f1  = (const float*)d_in[25];
  const float* W_f2  = (const float*)d_in[26];
  const float* b_f2  = (const float*)d_in[27];
  const float* W_h1  = (const float*)d_in[28];
  const float* b_h1  = (const float*)d_in[29];
  const float* W_h2  = (const float*)d_in[30];
  const float* b_h2  = (const float*)d_in[31];

  char* ws = (char*)d_ws;
  size_t off = 0;
  auto alloc = [&](size_t bytes) -> char* {
    char* p = ws + off;
    off = (off + bytes + 255) & ~(size_t)255;
    return p;
  };

  float*     embf    = (float*)    alloc((size_t)B_ * N_ * D_ * 4);
  _Float16*  h16     = (_Float16*) alloc((size_t)B_ * N_ * D_ * 2);
  _Float16*  ln1t16  = (_Float16*) alloc((size_t)B_ * D_ * N_ * 2);
  float*     qkvf    = (float*)    alloc((size_t)B_ * N_ * 3 * DI_ * 4);
  float*     gbuf    = (float*)    alloc((size_t)B_ * N_ * 3 * H_ * 4);
  _Float16*  Wqkv16  = (_Float16*) alloc((size_t)D_ * 3 * DI_ * 2);
  _Float16*  Wkc16   = (_Float16*) alloc((size_t)H_ * 256 * DH_ * 2);
  _Float16*  Wvc16   = (_Float16*) alloc((size_t)H_ * 256 * DH_ * 2);
  _Float16*  Wo16    = (_Float16*) alloc((size_t)DI_ * D_ * 2);
  _Float16*  Wt116   = (_Float16*) alloc((size_t)N_ * FF_ * 2);
  _Float16*  Wt216   = (_Float16*) alloc((size_t)FF_ * N_ * 2);
  _Float16*  Wf116   = (_Float16*) alloc((size_t)D_ * FF_ * 2);
  _Float16*  Wf216   = (_Float16*) alloc((size_t)FF_ * D_ * 2);
  _Float16*  kb16    = (_Float16*) alloc((size_t)H_ * B_ * NBLK_ * 256 * 2);
  _Float16*  vb16    = (_Float16*) alloc((size_t)H_ * B_ * NBLK_ * 256 * 2);
  float*     kcg     = (float*)    alloc((size_t)H_ * B_ * NBLK_ * DH_ * 4);
  float*     vcg     = (float*)    alloc((size_t)H_ * B_ * NBLK_ * DH_ * 4);
  float*     kcf     = (float*)    alloc((size_t)B_ * H_ * 17 * DH_ * 4);
  float*     vcf     = (float*)    alloc((size_t)B_ * H_ * 17 * DH_ * 4);
  _Float16*  o16     = (_Float16*) alloc((size_t)B_ * N_ * DI_ * 2);
  float*     attnout = (float*)    alloc((size_t)B_ * N_ * D_ * 4);
  _Float16*  t1_16   = (_Float16*) alloc((size_t)B_ * D_ * FF_ * 2);
  float*     t2f     = (float*)    alloc((size_t)B_ * D_ * N_ * 4);
  float*     ybuf    = (float*)    alloc((size_t)B_ * N_ * D_ * 4);
  _Float16*  fn16    = (_Float16*) alloc((size_t)B_ * N_ * D_ * 2);
  _Float16*  f1_16   = (_Float16*) alloc((size_t)B_ * N_ * FF_ * 2);
  float*     ffbuf   = (float*)    alloc((size_t)B_ * N_ * D_ * 4);
  float*     pooled  = (float*)    alloc((size_t)B_ * D_ * 4);

  // ---- weight conversions to f16
  auto cvt = [&](const float* s, _Float16* d, int n) {
    k_cvt16<<<(n + 255) / 256, 256, 0, stream>>>(s, d, n);
  };
  cvt(W_qkv, Wqkv16, D_ * 3 * DI_);
  cvt(W_kc,  Wkc16,  H_ * 256 * DH_);
  cvt(W_vc,  Wvc16,  H_ * 256 * DH_);
  cvt(W_o,   Wo16,   DI_ * D_);
  cvt(W_t1,  Wt116,  N_ * FF_);
  cvt(W_t2,  Wt216,  FF_ * N_);
  cvt(W_f1,  Wf116,  D_ * FF_);
  cvt(W_f2,  Wf216,  FF_ * D_);

  // ---- embed + rmsnorm + ln1 transpose
  k_embed<<<B_ * N_, D_, 0, stream>>>(x, W_emb, b_emb, rms_g, ln1_g, ln1_b,
                                      embf, h16, ln1t16);

  // ---- gates
  {
    int tot = B_ * N_ * 3 * H_;
    k_gates<<<(tot + 255) / 256, 256, 0, stream>>>(h16, W_g, b_g, gbuf);
  }

  // ---- qkv = h @ W_qkv  (8192 x 64) x (64 x 384)
  k_wmma_gemm_t64<<<dim3(128, 6, 1), 128, 0, stream>>>(h16, Wqkv16, qkvf, nullptr, nullptr,
                                                       8192, 64, 384, 0ULL, 0ULL, 0ULL, 0);

  // ---- kb/vb tiles
  {
    int tot = H_ * B_ * NBLK_ * 256;
    k_build_kvb<<<(tot + 255) / 256, 256, 0, stream>>>(qkvf, k_pos, v_pos, kb16, vb16);
  }

  // ---- per-head compression GEMMs: (512 x 256) x (256 x 16), batch = H
  k_wmma_gemm<<<dim3(32, 1, H_), 32, 0, stream>>>(
      kb16, Wkc16, kcg, 512, 256, 16,
      (unsigned long long)(512 * 256), (unsigned long long)(256 * 16),
      (unsigned long long)(512 * 16));
  k_wmma_gemm<<<dim3(32, 1, H_), 32, 0, stream>>>(
      vb16, Wvc16, vcg, 512, 256, 16,
      (unsigned long long)(512 * 256), (unsigned long long)(256 * 16),
      (unsigned long long)(512 * 16));

  // ---- bias + memory-token prepend
  {
    int tot = B_ * H_ * 17 * DH_;
    k_finish_kv<<<(tot + 255) / 256, 256, 0, stream>>>(kcg, vcg, b_kc, b_vc,
                                                       mem_k, mem_v, kcf, vcf);
  }

  // ---- fused attention (compressed + selected + sliding, gated merge)
  {
    int tot = B_ * H_ * N_;
    k_attn<<<(tot + 255) / 256, 256, 0, stream>>>(qkvf, kcf, vcf, gbuf, o16);
  }

  // ---- attn_out = o @ W_o  (8192 x 128) x (128 x 64)
  k_wmma_gemm_t64<<<dim3(128, 1, 1), 128, 0, stream>>>(o16, Wo16, attnout, nullptr, nullptr,
                                                       8192, 128, 64, 0ULL, 0ULL, 0ULL, 0);

  // ---- token-mixing branch: (2048 x 256) x (256 x 256) x2
  k_wmma_gemm_t64<<<dim3(32, 4, 1), 128, 0, stream>>>(ln1t16, Wt116, nullptr, t1_16, b_t1,
                                                      2048, 256, 256, 0ULL, 0ULL, 0ULL, 1);
  k_wmma_gemm_t64<<<dim3(32, 4, 1), 128, 0, stream>>>(t1_16, Wt216, t2f, nullptr, b_t2,
                                                      2048, 256, 256, 0ULL, 0ULL, 0ULL, 2);

  // ---- y = emb + t^T ; f = LN(y)
  k_make_y<<<B_ * N_, D_, 0, stream>>>(embf, t2f, ln2_g, ln2_b, ybuf, fn16);

  // ---- channel FF: (8192 x 64) x (64 x 256) gelu; (8192 x 256) x (256 x 64)
  k_wmma_gemm_t64<<<dim3(128, 4, 1), 128, 0, stream>>>(fn16, Wf116, nullptr, f1_16, b_f1,
                                                       8192, 64, 256, 0ULL, 0ULL, 0ULL, 1);
  k_wmma_gemm_t64<<<dim3(128, 1, 1), 128, 0, stream>>>(f1_16, Wf216, ffbuf, nullptr, b_f2,
                                                       8192, 256, 64, 0ULL, 0ULL, 0ULL, 2);

  // ---- pool + head
  {
    int tot = B_ * D_;
    k_pool<<<(tot + 255) / 256, 256, 0, stream>>>(ybuf, ffbuf, attnout, pooled);
  }
  k_head<<<B_, 32, 0, stream>>>(pooled, W_h1, b_h1, W_h2, b_h2, (float*)d_out);
}
